// GraphNeuralNetwork_57458072485900
// MI455X (gfx1250) — matrix operand order repaired and checked by reading, not compiled, not run on hardware
//
#include <hip/hip_runtime.h>

// ---------------------------------------------------------------------------
// GNN edge-layer network on MI455X (gfx1250).
// Dense-materialization strategy: per layer, scatter the E sparse edge weights
// into a dense f16 W^T [dst][src] matrix with global_atomic_pk_add_f16 (exact
// duplicate-edge accumulation), then X_{l+1} = relu(X_l * W + b) as a dense
// GEMM on v_wmma_f32_16x16x32_f16. W^T in f16 is 128MB -> the whole working
// set (W + f16 activations + edge lists) is resident in the 192MB L2, so the
// GEMM is WMMA-bound, not HBM-bound. The per-block B slab is staged into LDS
// with GLOBAL_LOAD_ASYNC_TO_LDS_B64 (double-buffered, ASYNCcnt-tracked).
// Activations are scaled 16^l before f16 conversion (per-layer ~16x decay:
// w_scale = sqrt(2/16384), fan-in 32) and unscaled in the fp32 epilogue.
// After each layer edges are subtracted back out (cheaper than 128MB memset;
// residues are ~1 f16 ulp on the ~512 duplicate slots only -> negligible).
// Fallback sparse-atomic path if the workspace is too small for dense W.
// ---------------------------------------------------------------------------

typedef __attribute__((ext_vector_type(16))) _Float16 v16h;
typedef __attribute__((ext_vector_type(8)))  _Float16 v8h;
typedef __attribute__((ext_vector_type(8)))  float    v8f;

// ---------------- common small kernels ----------------

// out16[i] = (f16)(in[i] * scale)
__global__ void k_cvt_f32_f16(const float* __restrict__ in,
                              _Float16* __restrict__ out,
                              float scale, int n) {
  int i = blockIdx.x * blockDim.x + threadIdx.x;
  if (i < n) out[i] = (_Float16)(in[i] * scale);
}

// W16[dst*N + src] += sign * w  (f16 packed atomic; other half adds +0.0)
__global__ void k_edge_scatter_f16(_Float16* __restrict__ W,
                                   const int* __restrict__ src,
                                   const int* __restrict__ dst,
                                   const float* __restrict__ w,
                                   int E, int N, float sign) {
  int e = blockIdx.x * blockDim.x + threadIdx.x;
  if (e < E) {
    const size_t idx = (size_t)dst[e] * N + src[e];
    union { unsigned short us[2]; unsigned u; } pk;
    pk.u = 0;
    pk.us[idx & 1] =
        __builtin_bit_cast(unsigned short, (_Float16)(sign * w[e]));
    _Float16* addr = W + (idx & ~(size_t)1);
    asm volatile("global_atomic_pk_add_f16 %0, %1, off scope:SCOPE_DEV"
                 :: "v"(addr), "v"(pk.u) : "memory");
  }
}

// ---------------- WMMA GEMM: [M,K]f16 x W^T[N,K]f16 -> [M,N] ---------------
// Block: 256 threads = 8 waves. Block owns a 32-wide n-strip over ALL of M
// (M=256 -> 16 m-tiles; each wave: 2 m-tiles x 2 n-tiles = 4 WMMAs/k-step),
// so W is read exactly once per layer (from L2). The 32n x 32k f16 B slab is
// async-copied global->LDS (B64 per thread), double-buffered so the copy of
// step k+1 overlaps the WMMAs of step k.

#define GEMM_BN   32   // n-width per block
#define BS_STRIDE 36   // LDS row stride in halves: 72B (8B-aligned async B64,
                       // 18-dword stride -> 16 distinct bank groups per frag)

__global__ void __launch_bounds__(256)
k_gemm_wmma(const _Float16* __restrict__ A,   // [M,K] f16 row-major (scaled act)
            const _Float16* __restrict__ Wh,  // [N,K] f16 row-major (dst-major)
            const float*    __restrict__ bias,// [N]
            float invS, float Snext,
            _Float16* __restrict__ Xnext,     // [M,N] f16 (when !last)
            float*    __restrict__ Out,       // [M,N] f32 (when last)
            int M, int N, int K, int last) {
  __shared__ _Float16 Bs[2][GEMM_BN * BS_STRIDE];

  const int tid  = threadIdx.x;
  const int lane = tid & 31;
  const int wid  = tid >> 5;          // 0..7
  const int lr   = lane & 15;
  const int hi   = lane >> 4;         // 0/1: half-wave id
  const int n0   = blockIdx.x * GEMM_BN;
  const int m0   = wid * 32;          // wave's 2 m-tiles: m0, m0+16

  // A rows for this lane's two m-tiles
  const _Float16* arow0 = A + (size_t)(m0 + lr) * K;
  const _Float16* arow1 = A + (size_t)(m0 + 16 + lr) * K;

  // B staging: thread t async-copies 8B: row (n0 + t/8), halves k0+(t%8)*4..+3
  const int sb_n = tid >> 3;
  const int sb_k = (tid & 7) * 4;
  const _Float16* gsrc = Wh + (size_t)(n0 + sb_n) * K + sb_k;
  const unsigned lds0 =
      (unsigned)(uintptr_t)&Bs[0][sb_n * BS_STRIDE + sb_k];
  const unsigned lds1 =
      (unsigned)(uintptr_t)&Bs[1][sb_n * BS_STRIDE + sb_k];

  v8f c00 = {}, c01 = {}, c10 = {}, c11 = {};

  // prologue: stage k0 = 0 into buffer 0
  asm volatile("global_load_async_to_lds_b64 %0, %1, off"
               :: "v"(lds0), "v"(gsrc) : "memory");
  asm volatile("s_wait_asynccnt 0x0" ::: "memory");
  __syncthreads();

  int cur = 0;
  for (int k0 = 0; k0 < K; k0 += 32) {
    // async-stage next slab into the other buffer (overlaps WMMAs below)
    if (k0 + 32 < K) {
      const unsigned ldsn = cur ? lds0 : lds1;
      asm volatile("global_load_async_to_lds_b64 %0, %1, off"
                   :: "v"(ldsn), "v"(gsrc + k0 + 32) : "memory");
    }

    // ---- B fragments (32x16 f16): lane lr -> col n; lanes0-15 K{0..7,16..23},
    //      lanes16-31 K{8..15,24..31} (matches A-matrix table order) ----
    const _Float16* bsc = &Bs[cur][0];
    v16h b0, b1;
    {
      const _Float16* bp0 = bsc + lr * BS_STRIDE + hi * 8;
      const _Float16* bp1 = bsc + (16 + lr) * BS_STRIDE + hi * 8;
#pragma unroll
      for (int i = 0; i < 16; ++i) { b0[i] = bp0[i + (i & 8)]; b1[i] = bp1[i + (i & 8)]; }
    }

    // ---- A fragments (16x32 f16): lanes0-15 K{0..7,16..23},
    //      lanes16-31 K{8..15,24..31} (per ISA A-matrix table) ----
    v16h a0, a1;
    {
      v8h l0 = *(const v8h*)(arow0 + k0 + hi * 8);
      v8h h0 = *(const v8h*)(arow0 + k0 + 16 + hi * 8);
      v8h l1 = *(const v8h*)(arow1 + k0 + hi * 8);
      v8h h1 = *(const v8h*)(arow1 + k0 + 16 + hi * 8);
#pragma unroll
      for (int i = 0; i < 8; ++i) {
        a0[i] = l0[i]; a0[8 + i] = h0[i];
        a1[i] = l1[i]; a1[8 + i] = h1[i];
      }
    }

    c00 = __builtin_amdgcn_wmma_f32_16x16x32_f16(false, a0, false, b0, (short)0, c00, false, false);
    c01 = __builtin_amdgcn_wmma_f32_16x16x32_f16(false, a0, false, b1, (short)0, c01, false, false);
    c10 = __builtin_amdgcn_wmma_f32_16x16x32_f16(false, a1, false, b0, (short)0, c10, false, false);
    c11 = __builtin_amdgcn_wmma_f32_16x16x32_f16(false, a1, false, b1, (short)0, c11, false, false);

    // drain this wave's async copy, then block-wide barrier before swap
    asm volatile("s_wait_asynccnt 0x0" ::: "memory");
    __syncthreads();
    cur ^= 1;
  }

  // ---- epilogue: D layout lane lr -> col n; VGPR r -> row m0 + r + 8*hi ----
  const v8f* cs[2][2] = { { &c00, &c01 }, { &c10, &c11 } };
#pragma unroll
  for (int mi = 0; mi < 2; ++mi) {
#pragma unroll
    for (int ni = 0; ni < 2; ++ni) {
      const int n  = n0 + ni * 16 + lr;
      const float bn = bias[n];
      const v8f cv = *cs[mi][ni];
#pragma unroll
      for (int r = 0; r < 8; ++r) {
        const int m = m0 + mi * 16 + r + hi * 8;
        float o = cv[r] * invS + bn;
        o = o > 0.f ? o : 0.f;
        if (last) Out[(size_t)m * N + n] = o;
        else      Xnext[(size_t)m * N + n] = (_Float16)(o * Snext);
      }
    }
  }
}

// ---------------- sparse fallback ([N,B] layout) ----------------

__global__ void k_transpose(const float* __restrict__ in, float* __restrict__ out,
                            int R, int C) {  // out[c*R+r] = in[r*C+c]
  __shared__ float tile[32][33];
  const int c0 = blockIdx.x * 32, r0 = blockIdx.y * 32;
  const int tx = threadIdx.x, ty = threadIdx.y;
  for (int i = 0; i < 32; i += 8) {
    int r = r0 + ty + i, c = c0 + tx;
    if (r < R && c < C) tile[ty + i][tx] = in[(size_t)r * C + c];
  }
  __syncthreads();
  for (int i = 0; i < 32; i += 8) {
    int r = r0 + tx, c = c0 + ty + i;
    if (r < R && c < C) out[(size_t)c * R + r] = tile[tx][ty + i];
  }
}

__global__ void k_edge_sparse(const float* __restrict__ T,   // [N,B]
                              float* __restrict__ ACC,       // [N,B]
                              const int* __restrict__ src,
                              const int* __restrict__ dst,
                              const float* __restrict__ w,
                              int E, int B) {
  for (int e = blockIdx.x; e < E; e += gridDim.x) {
    const int s = src[e], d = dst[e];
    const float ww = w[e];
    const float* ts = T + (size_t)s * B;
    float* ad = ACC + (size_t)d * B;
    for (int b = threadIdx.x; b < B; b += blockDim.x)
      atomicAdd(&ad[b], ww * ts[b]);
  }
}

__global__ void k_bias_relu_T(const float* __restrict__ ACC,
                              const float* __restrict__ bias,
                              float* __restrict__ Tout, int N, int B) {
  int i = blockIdx.x * blockDim.x + threadIdx.x;
  if (i < N * B) {
    float v = ACC[i] + bias[i / B];
    Tout[i] = v > 0.f ? v : 0.f;
  }
}

// ---------------- launcher ----------------

extern "C" void kernel_launch(void* const* d_in, const int* in_sizes, int n_in,
                              void* d_out, int out_size, void* d_ws, size_t ws_size,
                              hipStream_t stream) {
  const float* x    = (const float*)d_in[0];
  const float* wts  = (const float*)d_in[1];
  const float* bias = (const float*)d_in[2];
  const int*   esrc = (const int*)d_in[3];
  const int*   edst = (const int*)d_in[4];

  const int L = 4;
  const int N = in_sizes[2] / L;        // bias is [L,N]
  const int B = in_sizes[0] / N;        // x is [B,N]
  const int E = in_sizes[3] / L;        // edge_src is [L,E]
  const size_t MN = (size_t)B * N;

  const size_t wBytes  = (size_t)N * N * sizeof(_Float16);   // f16 W^T: 128MB
  const size_t xhBytes = ((MN * sizeof(_Float16)) + 255) & ~(size_t)255;
  const size_t denseNeed = wBytes + 2 * xhBytes;

  if (ws_size >= denseNeed && (B % 32) == 0 && (N % GEMM_BN) == 0) {
    // ---------------- dense WMMA path (L2-resident) ----------------
    _Float16* W  = (_Float16*)d_ws;
    _Float16* Xa = (_Float16*)((char*)d_ws + wBytes);
    _Float16* Xb = (_Float16*)((char*)Xa + xhBytes);

    hipMemsetAsync(W, 0, wBytes, stream);  // once per call (ws arrives poisoned)

    k_cvt_f32_f16<<<(int)((MN + 255) / 256), 256, 0, stream>>>(x, Xa, 1.0f, (int)MN);

    const float STEP = 16.0f;   // ~per-layer activation decay compensation
    float S = 1.0f;
    _Float16* cur = Xa; _Float16* nxt = Xb;
    const int gemmBlocks = N / GEMM_BN;
    const int scatBlocks = (E + 255) / 256;

    for (int l = 0; l < L; ++l) {
      const int last = (l == L - 1);
      k_edge_scatter_f16<<<scatBlocks, 256, 0, stream>>>(
          W, esrc + (size_t)l * E, edst + (size_t)l * E, wts + (size_t)l * E,
          E, N, 1.0f);
      const float Snext = S * STEP;
      k_gemm_wmma<<<gemmBlocks, 256, 0, stream>>>(
          cur, W, bias + (size_t)l * N, 1.0f / S, Snext,
          nxt, (float*)d_out, B, N, N, last);
      if (!last) {
        k_edge_scatter_f16<<<scatBlocks, 256, 0, stream>>>(  // un-build -> ~0
            W, esrc + (size_t)l * E, edst + (size_t)l * E, wts + (size_t)l * E,
            E, N, -1.0f);
      }
      S = Snext;
      _Float16* t = cur; cur = nxt; nxt = t;
    }
  } else {
    // ---------------- sparse fallback (24MB ws) ----------------
    float* Ta  = (float*)d_ws;
    float* Tb  = Ta + MN;
    float* ACC = Tb + MN;

    dim3 tb(32, 8);
    dim3 tgF((N + 31) / 32, (B + 31) / 32);  // x [B,N] -> Ta [N,B]
    k_transpose<<<tgF, tb, 0, stream>>>(x, Ta, B, N);

    float* cur = Ta; float* nxt = Tb;
    for (int l = 0; l < L; ++l) {
      hipMemsetAsync(ACC, 0, MN * sizeof(float), stream);
      k_edge_sparse<<<4096, 256, 0, stream>>>(
          cur, ACC, esrc + (size_t)l * E, edst + (size_t)l * E,
          wts + (size_t)l * E, E, B);
      k_bias_relu_T<<<(int)((MN + 255) / 256), 256, 0, stream>>>(
          ACC, bias + (size_t)l * N, nxt, N, B);
      float* t = cur; cur = nxt; nxt = t;
    }
    dim3 tgB((B + 31) / 32, (N + 31) / 32);  // cur [N,B] -> d_out [B,N]
    k_transpose<<<tgB, tb, 0, stream>>>(cur, (float*)d_out, N, B);
  }
}
